// RIM_17523466567974
// MI455X (gfx1250) — matrix operand order repaired and checked by reading, not compiled, not run on hardware
//
#include <hip/hip_runtime.h>
#include <math.h>

typedef __attribute__((ext_vector_type(16))) _Float16 v16h;
typedef __attribute__((ext_vector_type(8)))  _Float16 v8h;
typedef __attribute__((ext_vector_type(8)))  float    v8f;

constexpr int cB   = 256;
constexpr int cT   = 128;
constexpr int cIN  = 1024;
constexpr int cNR  = 6;
constexpr int cNH  = 4;
constexpr int cAT  = 128;
constexpr int cKS  = 256;   // 64*NH
constexpr int cVS  = 128;   // 32*NH
constexpr int cCKS = 128;   // 32*NH
constexpr int cOUT = 512;
constexpr int cTK  = 4;

// ---------------------------------------------------------------------------
// LDS-staged WMMA GEMM: C[M,N] = (A[M,K] @ op(B)) + bias (+C if ACCUM), opt ReLU
// Block (256 thr, 8 waves) computes a 64x64 macro-tile:
//   - per 32-wide k-step, threads cooperatively stage A(64x32) and B(64x32,
//     n-major) into LDS as f16 (8 KB total of the WGP's 320 KB)
//   - each wave owns one 16-row M tile and two 16-col N tiles: A fragment is
//     read once from LDS (2x v8h -> ds_load_b128) and reused for 2 WMMAs
// Fragment layouts (A/B use this code's contiguous k order, not the ISA table's; C/D per ISA):
//   A 16x32 f16: lane(half,sub) -> row m=sub, k = half*16 + i (contiguous in LDS)
//   B 32x16 f16: lane holds col n=sub, k = half*16 + i  (contiguous in LDS)
//   C/D: vgpr r -> row = r + 8*half, col = sub
// Requires M%64==0, N%64==0, K%32==0 (true for every GEMM in this net).
// ---------------------------------------------------------------------------
template<bool TRANSB, bool ACCUM, bool RELU>
__global__ __launch_bounds__(256)
void gemm_wmma(const float* __restrict__ A, const float* __restrict__ Bm,
               const float* __restrict__ bias, float* __restrict__ C,
               int M, int N, int K, int lda, int ldb, int ldc,
               long sAz, long sBz, long sbz, long sCz)
{
    constexpr int KT = 32;
    __shared__ __align__(32) _Float16 As[64 * KT];   // [m][k]
    __shared__ __align__(32) _Float16 Bs[64 * KT];   // [n][k]

    int z = blockIdx.y;
    A    += (long)z * sAz;
    Bm   += (long)z * sBz;
    bias += (long)z * sbz;
    C    += (long)z * sCz;

    int tid = threadIdx.x;
    int macrosN = N >> 6;
    int bM = (blockIdx.x / macrosN) << 6;
    int bN = (blockIdx.x % macrosN) << 6;

    int wave = tid >> 5, lane = tid & 31;
    int wm = wave & 3;              // M tile 0..3
    int wn = (wave >> 2) << 1;      // N tiles {wn, wn+1}, wn in {0,2}
    int half = lane >> 4, sub = lane & 15;

    // staging assignments
    int ar = tid >> 2;              // 0..63 row (A and transposed-B)
    int ac = (tid & 3) << 3;        // 0,8,16,24
    int bk = tid >> 3;              // 0..31 k-row (non-trans B)
    int bn = (tid & 7) << 3;        // 0..56 n-col

    v8f acc0 = {}, acc1 = {};

    for (int k0 = 0; k0 < K; k0 += KT) {
        // ---- stage A (64x32) ----
        {
            const float* ap = A + (long)(bM + ar) * lda + k0 + ac;
            float4 x0 = ((const float4*)ap)[0];
            float4 x1 = ((const float4*)ap)[1];
            _Float16* d = As + ar * KT + ac;
            d[0] = (_Float16)x0.x; d[1] = (_Float16)x0.y;
            d[2] = (_Float16)x0.z; d[3] = (_Float16)x0.w;
            d[4] = (_Float16)x1.x; d[5] = (_Float16)x1.y;
            d[6] = (_Float16)x1.z; d[7] = (_Float16)x1.w;
            if (k0 + KT < K)
                __builtin_prefetch(ap + KT, 0, 1);   // global_prefetch_b8
        }
        // ---- stage B (64 n x 32 k, n-major in LDS) ----
        if (TRANSB) {
            // B stored (N,K): row n is contiguous in k
            const float* bp = Bm + (long)(bN + ar) * ldb + k0 + ac;
            float4 y0 = ((const float4*)bp)[0];
            float4 y1 = ((const float4*)bp)[1];
            _Float16* d = Bs + ar * KT + ac;
            d[0] = (_Float16)y0.x; d[1] = (_Float16)y0.y;
            d[2] = (_Float16)y0.z; d[3] = (_Float16)y0.w;
            d[4] = (_Float16)y1.x; d[5] = (_Float16)y1.y;
            d[6] = (_Float16)y1.z; d[7] = (_Float16)y1.w;
            if (k0 + KT < K)
                __builtin_prefetch(bp + KT, 0, 1);
        } else {
            // B stored (K,N): coalesced load along n, transpose into LDS
            const float* bp = Bm + (long)(k0 + bk) * ldb + bN + bn;
            float4 y0 = ((const float4*)bp)[0];
            float4 y1 = ((const float4*)bp)[1];
            _Float16 v[8] = { (_Float16)y0.x, (_Float16)y0.y, (_Float16)y0.z, (_Float16)y0.w,
                              (_Float16)y1.x, (_Float16)y1.y, (_Float16)y1.z, (_Float16)y1.w };
            #pragma unroll
            for (int i = 0; i < 8; ++i)
                Bs[(bn + i) * KT + bk] = v[i];
            if (k0 + KT < K)
                __builtin_prefetch(Bm + (long)(k0 + KT + bk) * ldb + bN + bn, 0, 1);
        }
        __syncthreads();

        // ---- fragments from LDS ----
        v16h a;
        {
            const _Float16* arow = As + (wm * 16 + sub) * KT;
            a = *(const v16h*)(arow + half * 16);
        }
        v16h b0 = *(const v16h*)(Bs + (wn * 16 + sub) * KT + half * 16);
        v16h b1 = *(const v16h*)(Bs + ((wn + 1) * 16 + sub) * KT + half * 16);

        acc0 = __builtin_amdgcn_wmma_f32_16x16x32_f16(
            false, a, false, b0, (short)0, acc0, false, false);
        acc1 = __builtin_amdgcn_wmma_f32_16x16x32_f16(
            false, a, false, b1, (short)0, acc1, false, false);
        __syncthreads();
    }

    // ---- epilogue ----
    int col0 = bN + wn * 16 + sub;
    int col1 = col0 + 16;
    float bv0 = bias[col0];
    float bv1 = bias[col1];
    #pragma unroll
    for (int r = 0; r < 8; ++r) {
        long row = bM + wm * 16 + r + 8 * half;
        float u0 = acc0[r] + bv0;
        float u1 = acc1[r] + bv1;
        if (ACCUM) { u0 += C[row * (long)ldc + col0]; u1 += C[row * (long)ldc + col1]; }
        if (RELU)  { u0 = fmaxf(u0, 0.f); u1 = fmaxf(u1, 0.f); }
        C[row * (long)ldc + col0] = u0;
        C[row * (long)ldc + col1] = u1;
    }
}

template<bool TB, bool AC, bool RE>
static inline void launch_gemm(hipStream_t st, const float* A, const float* Bm,
                               const float* bias, float* C,
                               int M, int N, int K, int lda, int ldb, int ldc,
                               long sA, long sB, long sb, long sC, int nz)
{
    dim3 grid((M / 64) * (N / 64), nz, 1);
    gemm_wmma<TB, AC, RE><<<grid, 256, 0, st>>>(A, Bm, bias, C, M, N, K,
                                                lda, ldb, ldc, sA, sB, sb, sC);
}

// ---------------------------------------------------------------------------
// LayerNorm over last-two-dims flattened to length L, per batch row.
// Optional second input (residual). gamma/beta of length L.
// ---------------------------------------------------------------------------
__global__ __launch_bounds__(256)
void ln2d_kernel(const float* __restrict__ x1, const float* __restrict__ x2,
                 const float* __restrict__ g, const float* __restrict__ bta,
                 float* __restrict__ out, int L)
{
    long base = (long)blockIdx.x * L;
    float s = 0.f, q = 0.f;
    for (int i = threadIdx.x; i < L; i += 256) {
        float v = x1[base + i] + (x2 ? x2[base + i] : 0.f);
        s += v; q += v * v;
    }
    for (int o = 16; o > 0; o >>= 1) {
        s += __shfl_xor(s, o, 32);
        q += __shfl_xor(q, o, 32);
    }
    __shared__ float rs[8], rq[8];
    if ((threadIdx.x & 31) == 0) { rs[threadIdx.x >> 5] = s; rq[threadIdx.x >> 5] = q; }
    __syncthreads();
    float S = 0.f, Q = 0.f;
    #pragma unroll
    for (int w = 0; w < 8; ++w) { S += rs[w]; Q += rq[w]; }
    float inv  = 1.f / (float)L;
    float mean = S * inv;
    float var  = Q * inv - mean * mean;
    float rr   = rsqrtf(var + 1e-5f);
    for (int i = threadIdx.x; i < L; i += 256) {
        float v = x1[base + i] + (x2 ? x2[base + i] : 0.f);
        out[base + i] = (v - mean) * rr * g[i] + bta[i];
    }
}

// dst[b*stride + j] = bias[j]  (fills the "null token" rows with bias)
__global__ void fill_rows(float* __restrict__ dst, const float* __restrict__ bias,
                          int n, int stride, int total)
{
    int idx = blockIdx.x * 256 + threadIdx.x;
    if (idx >= total) return;
    int b = idx / n, j = idx - b * n;
    dst[(long)b * stride + j] = bias[j];
}

__global__ void zero_kernel(float* __restrict__ p, int n)
{
    int i = blockIdx.x * 256 + threadIdx.x;
    if (i < n) p[i] = 0.f;
}

// RIM input attention: NR=6 queries vs S1=2 keys, NH=4 heads, qd=64, vd=32.
__global__ void attn_rim(const float* __restrict__ Q, const float* __restrict__ Kn,
                         const float* __restrict__ Vn, float* __restrict__ att,
                         float* __restrict__ w0buf)
{
    int idx = blockIdx.x * 256 + threadIdx.x;
    if (idx >= cB * cNH * cNR) return;
    int r = idx % cNR;
    int h = (idx / cNR) % cNH;
    int b = idx / (cNH * cNR);
    const float sc = 0.0625f;                    // KS^-0.5 = 256^-0.5
    const float* q  = Q  + (long)b * (cNR * cKS) + r * cKS + h * 64;
    const float* k0 = Kn + (long)b * (2 * cKS) + h * 64;
    const float* k1 = k0 + cKS;
    float d0 = 0.f, d1 = 0.f;
    #pragma unroll 8
    for (int j = 0; j < 64; ++j) { float qv = q[j] * sc; d0 += qv * k0[j]; d1 += qv * k1[j]; }
    float mx = fmaxf(d0, d1);
    float e0 = expf(d0 - mx), e1 = expf(d1 - mx);
    float inv = 1.f / (e0 + e1);
    float w0 = e0 * inv, w1 = e1 * inv;
    w0buf[(long)b * cNH * cNR + h * cNR + r] = w0;
    const float* v0 = Vn + (long)b * (2 * cVS) + h * 32;
    const float* v1 = v0 + cVS;
    float* o = att + (long)b * (cNR * cVS) + r * cVS + h * 32;
    #pragma unroll 8
    for (int j = 0; j < 32; ++j) o[j] = w0 * v0[j] + w1 * v1[j];
}

// null_w = mean over heads of w[:, :, :, 0]; mask = (null_w <= 4th smallest)
__global__ void mask_topk(const float* __restrict__ w0, float* __restrict__ mask)
{
    int b = blockIdx.x * 256 + threadIdx.x;
    if (b >= cB) return;
    float nw[cNR];
    for (int r = 0; r < cNR; ++r) {
        float s = 0.f;
        for (int h = 0; h < cNH; ++h) s += w0[(long)b * cNH * cNR + h * cNR + r];
        nw[r] = s * (1.f / cNH);
    }
    float srt[cNR];
    for (int r = 0; r < cNR; ++r) srt[r] = nw[r];
    for (int i = 1; i < cNR; ++i) {
        float key = srt[i]; int j = i - 1;
        while (j >= 0 && srt[j] > key) { srt[j + 1] = srt[j]; --j; }
        srt[j + 1] = key;
    }
    float kth = srt[cTK - 1];
    for (int r = 0; r < cNR; ++r) mask[b * cNR + r] = (nw[r] <= kth) ? 1.f : 0.f;
}

// LSTM cell update with per-module mask blending
__global__ void lstm_update(const float* __restrict__ gts, const float* __restrict__ hx,
                            const float* __restrict__ cx, const float* __restrict__ mask,
                            float* __restrict__ h2, float* __restrict__ cx2)
{
    int idx = blockIdx.x * 256 + threadIdx.x;
    if (idx >= cB * cNR * cAT) return;
    int j = idx % cAT;
    int r = (idx / cAT) % cNR;
    int b = idx / (cNR * cAT);
    const float* g = gts + (long)b * (cNR * 4 * cAT) + r * 4 * cAT;
    float ig = 1.f / (1.f + expf(-g[j]));
    float fg = 1.f / (1.f + expf(-g[cAT + j]));
    float gg = tanhf(g[2 * cAT + j]);
    float og = 1.f / (1.f + expf(-g[3 * cAT + j]));
    float cold = cx[idx];
    float cnew = fg * cold + ig * gg;
    float hnew = og * tanhf(cnew);
    float m = mask[b * cNR + r];
    h2[idx]  = hx[idx] * (1.f - m) + hnew * m;
    cx2[idx] = cold    * (1.f - m) + cnew * m;
}

// Communication attention: 6 queries vs 6 keys, 4 heads, d=32
__global__ void comm_attn(const float* __restrict__ Q, const float* __restrict__ Kn,
                          const float* __restrict__ Vn, float* __restrict__ out)
{
    int idx = blockIdx.x * 256 + threadIdx.x;
    if (idx >= cB * cNH * cNR) return;
    int r = idx % cNR;
    int h = (idx / cNR) % cNH;
    int b = idx / (cNH * cNR);
    const float sc = 0.08838834764831845f;       // CKS^-0.5 = 128^-0.5
    const float* q = Q + (long)b * (cNR * cCKS) + r * cCKS + h * 32;
    float lg[cNR];
    float mx = -1e30f;
    for (int s = 0; s < cNR; ++s) {
        const float* k = Kn + (long)b * (cNR * cCKS) + s * cCKS + h * 32;
        float d = 0.f;
        #pragma unroll 8
        for (int j = 0; j < 32; ++j) d += q[j] * k[j];
        lg[s] = d * sc;
        mx = fmaxf(mx, lg[s]);
    }
    float den = 0.f;
    for (int s = 0; s < cNR; ++s) { lg[s] = expf(lg[s] - mx); den += lg[s]; }
    float inv = 1.f / den;
    float* o = out + (long)b * (cNR * cCKS) + r * cCKS + h * 32;
    for (int j = 0; j < 32; ++j) {
        float a = 0.f;
        for (int s = 0; s < cNR; ++s)
            a += lg[s] * Vn[(long)b * (cNR * cCKS) + s * cCKS + h * 32 + j];
        o[j] = a * inv;
    }
}

// hx3 = h_com*m + hx2*(1-m)
__global__ void blend_state(const float* __restrict__ hcom, const float* __restrict__ h2,
                            const float* __restrict__ mask, float* __restrict__ hx)
{
    int idx = blockIdx.x * 256 + threadIdx.x;
    if (idx >= cB * cNR * cAT) return;
    int r = (idx / cAT) % cNR;
    int b = idx / (cNR * cAT);
    float m = mask[b * cNR + r];
    hx[idx] = hcom[idx] * m + h2[idx] * (1.f - m);
}

// ---------------------------------------------------------------------------
extern "C" void kernel_launch(void* const* d_in, const int* in_sizes, int n_in,
                              void* d_out, int out_size, void* d_ws, size_t ws_size,
                              hipStream_t stream)
{
    (void)in_sizes; (void)n_in; (void)out_size; (void)ws_size;

    const float* inputs = (const float*)d_in[0];
    const float* We   = (const float*)d_in[1];  const float* be   = (const float*)d_in[2];
    const float* Wq   = (const float*)d_in[3];  const float* bq   = (const float*)d_in[4];
    const float* q_g  = (const float*)d_in[5];  const float* q_b  = (const float*)d_in[6];
    const float* Wk   = (const float*)d_in[7];  const float* bk   = (const float*)d_in[8];
    const float* k_g  = (const float*)d_in[9];  const float* k_b  = (const float*)d_in[10];
    const float* Wv   = (const float*)d_in[11]; const float* bv   = (const float*)d_in[12];
    const float* v_g  = (const float*)d_in[13]; const float* v_b  = (const float*)d_in[14];
    const float* m1W  = (const float*)d_in[15]; const float* m1b  = (const float*)d_in[16];
    const float* m2W  = (const float*)d_in[17]; const float* m2b  = (const float*)d_in[18];
    const float* a1_g = (const float*)d_in[19]; const float* a1_b = (const float*)d_in[20];
    const float* a2_g = (const float*)d_in[21]; const float* a2_b = (const float*)d_in[22];
    const float* Wih  = (const float*)d_in[23]; const float* Whh  = (const float*)d_in[24];
    const float* bih  = (const float*)d_in[25]; const float* bhh  = (const float*)d_in[26];
    const float* cWq  = (const float*)d_in[27]; const float* cbq  = (const float*)d_in[28];
    const float* cq_g = (const float*)d_in[29]; const float* cq_b = (const float*)d_in[30];
    const float* cWk  = (const float*)d_in[31]; const float* cbk  = (const float*)d_in[32];
    const float* ck_g = (const float*)d_in[33]; const float* ck_b = (const float*)d_in[34];
    const float* cWv  = (const float*)d_in[35]; const float* cbv  = (const float*)d_in[36];
    const float* cv_g = (const float*)d_in[37]; const float* cv_b = (const float*)d_in[38];
    const float* cm1W = (const float*)d_in[39]; const float* cm1b = (const float*)d_in[40];
    const float* cm2W = (const float*)d_in[41]; const float* cm2b = (const float*)d_in[42];
    const float* c1_g = (const float*)d_in[43]; const float* c1_b = (const float*)d_in[44];
    const float* c2_g = (const float*)d_in[45]; const float* c2_b = (const float*)d_in[46];
    const float* Wo   = (const float*)d_in[47]; const float* bo   = (const float*)d_in[48];
    float* out = (float*)d_out;

    // Workspace layout (floats)
    float* p = (float*)d_ws;
    float* E    = p; p += cB * cAT;               // embed
    float* KL   = p; p += cB * 2 * cKS;           // [null; e@Wk] then LN
    float* VL   = p; p += cB * 2 * cVS;
    float* QL   = p; p += cB * cNR * cKS;
    float* W0   = p; p += cB * cNH * cNR;         // null-token attention weights
    float* MSK  = p; p += cB * cNR;
    float* ATT  = p; p += cB * cNR * cVS;         // att -> an (LN in place)
    float* M1   = p; p += cB * cNR * cAT;
    float* M2   = p; p += cB * cNR * cAT;
    float* HATT = p; p += cB * cNR * cAT;
    float* GTS  = p; p += cB * cNR * 4 * cAT;
    float* H2   = p; p += cB * cNR * cAT;         // hx2 (= h_old)
    float* CQ   = p; p += cB * cNR * cCKS;
    float* CK   = p; p += cB * cNR * cCKS;
    float* CVb  = p; p += cB * cNR * cCKS;
    float* CATT = p; p += cB * cNR * cCKS;        // catt -> cn (LN in place)
    float* CM1  = p; p += cB * cNR * cAT;
    float* CM2  = p; p += cB * cNR * cAT;
    float* HCOM = p; p += cB * cNR * cAT;
    float* HX   = p; p += cB * cNR * cAT;         // carry h
    float* CX   = p; p += cB * cNR * cAT;         // carry c (contiguous after HX)

    // zero recurrent state (HX and CX are contiguous)
    {
        int n = 2 * cB * cNR * cAT;
        zero_kernel<<<(n + 255) / 256, 256, 0, stream>>>(HX, n);
    }

    const int nSmall = cB * cNH * cNR;            // 6144
    const int nState = cB * cNR * cAT;            // 196608

    for (int t = 0; t < cT; ++t) {
        const float* Xt = inputs + (long)t * cIN;

        // e = x_t @ We + be
        launch_gemm<false, false, false>(stream, Xt, We, be, E,
            cB, cAT, cIN, cT * cIN, cAT, cAT, 0, 0, 0, 0, 1);

        // k = ln2d([0; e] @ Wk + bk) ; v likewise
        fill_rows<<<(cB * cKS + 255) / 256, 256, 0, stream>>>(KL, bk, cKS, 2 * cKS, cB * cKS);
        launch_gemm<false, false, false>(stream, E, Wk, bk, KL + cKS,
            cB, cKS, cAT, cAT, cKS, 2 * cKS, 0, 0, 0, 0, 1);
        fill_rows<<<(cB * cVS + 255) / 256, 256, 0, stream>>>(VL, bv, cVS, 2 * cVS, cB * cVS);
        launch_gemm<false, false, false>(stream, E, Wv, bv, VL + cVS,
            cB, cVS, cAT, cAT, cVS, 2 * cVS, 0, 0, 0, 0, 1);

        // q = ln2d(einsum(hx, Wq[r]) + bq[r])  (batched over r)
        launch_gemm<false, false, false>(stream, HX, Wq, bq, QL,
            cB, cKS, cAT, cNR * cAT, cKS, cNR * cKS,
            cAT, (long)cAT * cKS, cKS, cKS, cNR);

        ln2d_kernel<<<cB, 256, 0, stream>>>(KL, nullptr, k_g, k_b, KL, 2 * cKS);
        ln2d_kernel<<<cB, 256, 0, stream>>>(VL, nullptr, v_g, v_b, VL, 2 * cVS);
        ln2d_kernel<<<cB, 256, 0, stream>>>(QL, nullptr, q_g, q_b, QL, cNR * cKS);

        attn_rim<<<(nSmall + 255) / 256, 256, 0, stream>>>(QL, KL, VL, ATT, W0);
        mask_topk<<<(cB + 255) / 256, 256, 0, stream>>>(W0, MSK);

        // an = ln2d(att); mlp; h_attn = ln2d(an + m2)
        ln2d_kernel<<<cB, 256, 0, stream>>>(ATT, nullptr, a1_g, a1_b, ATT, cNR * cVS);
        launch_gemm<false, false, true>(stream, ATT, m1W, m1b, M1,
            cB * cNR, cAT, cAT, cAT, cAT, cAT, 0, 0, 0, 0, 1);
        launch_gemm<false, false, true>(stream, M1, m2W, m2b, M2,
            cB * cNR, cAT, cAT, cAT, cAT, cAT, 0, 0, 0, 0, 1);
        ln2d_kernel<<<cB, 256, 0, stream>>>(ATT, M2, a2_g, a2_b, HATT, cNR * cAT);

        // gates = h_attn @ Wih[r]^T + hx @ Whh[r]^T + bih + bhh
        launch_gemm<true, false, false>(stream, HATT, Wih, bih, GTS,
            cB, 4 * cAT, cAT, cNR * cAT, cAT, cNR * 4 * cAT,
            cAT, (long)4 * cAT * cAT, 4 * cAT, 4 * cAT, cNR);
        launch_gemm<true, true, false>(stream, HX, Whh, bhh, GTS,
            cB, 4 * cAT, cAT, cNR * cAT, cAT, cNR * 4 * cAT,
            cAT, (long)4 * cAT * cAT, 4 * cAT, 4 * cAT, cNR);

        lstm_update<<<(nState + 255) / 256, 256, 0, stream>>>(GTS, HX, CX, MSK, H2, CX);

        // communication projections (batched over r) + LN
        launch_gemm<false, false, false>(stream, H2, cWq, cbq, CQ,
            cB, cCKS, cAT, cNR * cAT, cCKS, cNR * cCKS,
            cAT, (long)cAT * cCKS, cCKS, cCKS, cNR);
        launch_gemm<false, false, false>(stream, H2, cWk, cbk, CK,
            cB, cCKS, cAT, cNR * cAT, cCKS, cNR * cCKS,
            cAT, (long)cAT * cCKS, cCKS, cCKS, cNR);
        launch_gemm<false, false, false>(stream, H2, cWv, cbv, CVb,
            cB, cCKS, cAT, cNR * cAT, cCKS, cNR * cCKS,
            cAT, (long)cAT * cCKS, cCKS, cCKS, cNR);
        ln2d_kernel<<<cB, 256, 0, stream>>>(CQ,  nullptr, cq_g, cq_b, CQ,  cNR * cCKS);
        ln2d_kernel<<<cB, 256, 0, stream>>>(CK,  nullptr, ck_g, ck_b, CK,  cNR * cCKS);
        ln2d_kernel<<<cB, 256, 0, stream>>>(CVb, nullptr, cv_g, cv_b, CVb, cNR * cCKS);

        comm_attn<<<(nSmall + 255) / 256, 256, 0, stream>>>(CQ, CK, CVb, CATT);

        // cn = ln2d(catt); cm = relu(relu(cn@cm1W+b)@cm2W+b); h_com = ln2d(hx2+cm)
        ln2d_kernel<<<cB, 256, 0, stream>>>(CATT, nullptr, c1_g, c1_b, CATT, cNR * cCKS);
        launch_gemm<false, false, true>(stream, CATT, cm1W, cm1b, CM1,
            cB * cNR, cAT, cCKS, cCKS, cAT, cAT, 0, 0, 0, 0, 1);
        launch_gemm<false, false, true>(stream, CM1, cm2W, cm2b, CM2,
            cB * cNR, cAT, cAT, cAT, cAT, cAT, 0, 0, 0, 0, 1);
        ln2d_kernel<<<cB, 256, 0, stream>>>(H2, CM2, c2_g, c2_b, HCOM, cNR * cAT);

        blend_state<<<(nState + 255) / 256, 256, 0, stream>>>(HCOM, H2, MSK, HX);

        // out[:, t, :] = hx3.reshape(B, NR*AT) @ Wo + bo
        launch_gemm<false, false, false>(stream, HX, Wo, bo, out + (long)t * cOUT,
            cB, cOUT, cNR * cAT, cNR * cAT, cOUT, cT * cOUT, 0, 0, 0, 0, 1);
    }
}
